// Layer_36584531427474
// MI455X (gfx1250) — matrix operand order repaired and checked by reading, not compiled, not run on hardware
//
#include <hip/hip_runtime.h>
#include <math.h>

// ---------------- problem constants (reference: B=4, L=2048, D=1024, DI=2048, K=4) --------
#define BB   4
#define LL   2048
#define DD   1024
#define DDI  2048
#define MM   (BB * LL)     // 8192 rows

typedef __attribute__((ext_vector_type(16))) __bf16 v16bf;  // WMMA A/B operand (8 VGPRs)
typedef __attribute__((ext_vector_type(8)))  float  v8f;    // WMMA C/D operand (8 VGPRs)

// ---------------- bf16 helpers (round-to-nearest-even) ------------------------------------
__device__ __forceinline__ __bf16 f2bf(float f) {
  union { float f; unsigned u; } c; c.f = f;
  unsigned r = c.u + 0x7fffu + ((c.u >> 16) & 1u);
  unsigned short h = (unsigned short)(r >> 16);
  __bf16 o; __builtin_memcpy(&o, &h, 2); return o;
}
__device__ __forceinline__ float bf2f(__bf16 b) {
  unsigned short h; __builtin_memcpy(&h, &b, 2);
  union { unsigned u; float f; } c; c.u = ((unsigned)h) << 16; return c.f;
}
__device__ __forceinline__ float silu_f(float x) { return x / (1.f + __expf(-x)); }

__device__ __forceinline__ v8f wmma_bf(v16bf a, v16bf b, v8f c) {
  // D(f32 16x16) = A(16x32 bf16) * B(32x16 bf16) + C
  return __builtin_amdgcn_wmma_f32_16x16x32_bf16(false, a, false, b, (short)0, c, false, false);
}

// ---------------- CDNA5 async global->LDS copy (ASYNCcnt path, ISA 15.18.3 op 98) ----------
__device__ __forceinline__ void async_b128(void* lds, const void* g) {
  unsigned l = (unsigned)(unsigned long long)lds;   // LDS aperture: low 32 bits = LDS offset
  asm volatile("global_load_async_to_lds_b128 %0, %1, off" :: "v"(l), "v"(g) : "memory");
}
__device__ __forceinline__ void wait_async0() {
  asm volatile("s_wait_asynccnt 0x0" ::: "memory");
}

// ---------------- fragment loaders per CDNA5 ISA VGPR layouts ------------------------------
__device__ __forceinline__ v16bf load_A_rm(const __bf16* base, int ld, int lane) {
  int row = lane & 15, kb = (lane >> 4) * 8;
  const __bf16* p = base + row * ld + kb;
  v16bf f;
#pragma unroll
  for (int e = 0; e < 8; ++e) f[e] = p[e];
#pragma unroll
  for (int e = 0; e < 8; ++e) f[e + 8] = p[e + 16];
  return f;
}
__device__ __forceinline__ v16bf load_A_rm_f32(const float* base, int ld, int lane) {
  int row = lane & 15, kb = (lane >> 4) * 8;
  const float* p = base + row * ld + kb;
  v16bf f;
#pragma unroll
  for (int e = 0; e < 8; ++e) f[e] = f2bf(p[e]);
#pragma unroll
  for (int e = 0; e < 8; ++e) f[e + 8] = f2bf(p[e + 16]);
  return f;
}
__device__ __forceinline__ v16bf load_B_ct(const __bf16* baseT, int ld, int lane) {
  int col = lane & 15, kb = (lane >> 4) * 8;
  const __bf16* p = baseT + col * ld + kb;
  v16bf f;
#pragma unroll
  for (int e = 0; e < 8; ++e) f[e] = p[e];
#pragma unroll
  for (int e = 0; e < 8; ++e) f[e + 8] = p[e + 16];
  return f;
}
__device__ __forceinline__ v16bf load_B_ct_f32(const float* baseT, int ld, int lane) {
  int col = lane & 15, kb = (lane >> 4) * 8;
  const float* p = baseT + col * ld + kb;
  v16bf f;
#pragma unroll
  for (int e = 0; e < 8; ++e) f[e] = f2bf(p[e]);
#pragma unroll
  for (int e = 0; e < 8; ++e) f[e + 8] = f2bf(p[e + 16]);
  return f;
}

// ---------------- fp32 -> bf16 transposed weight conversion (tiled) ------------------------
// in:  W  [K][N] fp32   out: Wt [N][K] bf16   (so GEMM B-staging is contiguous async b128)
__global__ __launch_bounds__(256) void cvt_t_kernel(const float* __restrict__ s,
                                                    __bf16* __restrict__ d, int K, int N) {
  __shared__ float tile[32][33];
  int n0 = blockIdx.x * 32, k0 = blockIdx.y * 32;
  int tx = threadIdx.x & 31, ty = threadIdx.x >> 5;
#pragma unroll
  for (int r = ty; r < 32; r += 8) tile[r][tx] = s[(size_t)(k0 + r) * N + n0 + tx];
  __syncthreads();
#pragma unroll
  for (int r = ty; r < 32; r += 8) d[(size_t)(n0 + r) * K + k0 + tx] = f2bf(tile[tx][r]);
}

// ---------------- per-batch bf16 transpose: [B][L][D] -> [B][D][L] -------------------------
__global__ __launch_bounds__(256) void transpose_bf16_kernel(const __bf16* __restrict__ s,
                                                             __bf16* __restrict__ d) {
  __shared__ __bf16 tile[32][33];
  int d0 = blockIdx.x * 32, l0 = blockIdx.y * 32, b = blockIdx.z;
  const __bf16* sb = s + (size_t)b * LL * DD;
  __bf16* db = d + (size_t)b * LL * DD;
  int tx = threadIdx.x & 31, ty = threadIdx.x >> 5;
#pragma unroll
  for (int r = ty; r < 32; r += 8) tile[r][tx] = sb[(size_t)(l0 + r) * DD + d0 + tx];
  __syncthreads();
#pragma unroll
  for (int r = ty; r < 32; r += 8) db[(size_t)(d0 + r) * LL + l0 + tx] = tile[tx][r];
}

// ---------------- RMSNorm: one workgroup per row -------------------------------------------
__global__ __launch_bounds__(256) void rmsnorm_kernel(const float* __restrict__ x,
                                                      const float* __restrict__ w,
                                                      __bf16* __restrict__ out) {
  __shared__ float red[256];
  int row = blockIdx.x, tid = threadIdx.x;
  const float* xr = x + (size_t)row * DD;
  float s = 0.f;
  for (int i = tid; i < DD; i += 256) { float v = xr[i]; s += v * v; }
  red[tid] = s; __syncthreads();
  for (int o = 128; o > 0; o >>= 1) { if (tid < o) red[tid] += red[tid + o]; __syncthreads(); }
  float sc = rsqrtf(red[0] / (float)DD + 1e-6f);
  for (int i = tid; i < DD; i += 256) out[(size_t)row * DD + i] = f2bf(xr[i] * sc * w[i]);
}

// ---------------- generic WMMA GEMM: out = act(A[M,K] @ Wt^T + bias) (+residual) -----------
// Wt is pre-transposed [N][K]. Block tile 128x128, 8 waves, 2 m-tiles x 4 n-tiles per wave.
// All staging is async global->LDS b128, double-buffered.
__global__ __launch_bounds__(256) void gemm_bf16_kernel(
    const __bf16* __restrict__ A, const __bf16* __restrict__ Wt,
    const float* __restrict__ bias, const float* __restrict__ residual,
    float* __restrict__ outF, __bf16* __restrict__ outB,
    int M, int K, int N, int act) {
  __shared__ __bf16 At[2][128 * 32];    // [row][k]
  __shared__ __bf16 Bt[2][128 * 32];    // [col][k]
  int tid = threadIdx.x, lane = tid & 31, wv = tid >> 5;
  int bm = blockIdx.x, bn = blockIdx.y;
  int mb = (wv & 3) * 2;       // first of 2 m-tiles
  int nb = (wv >> 2) * 4;      // first of 4 n-tiles
  v8f zero = {};
  v8f acc[2][4] = {{zero, zero, zero, zero}, {zero, zero, zero, zero}};

  int srow = tid >> 1, sk0 = (tid & 1) * 16;   // 32B per thread for each tile
  const __bf16* aBase = A  + (size_t)(bm * 128 + srow) * K + sk0;
  const __bf16* wBase = Wt + (size_t)(bn * 128 + srow) * K + sk0;
  int ldsOff = srow * 32 + sk0;

  // first stage
  async_b128(&At[0][ldsOff], aBase);
  async_b128(&At[0][ldsOff + 8], aBase + 8);
  async_b128(&Bt[0][ldsOff], wBase);
  async_b128(&Bt[0][ldsOff + 8], wBase + 8);

  int nsteps = K / 32;
  for (int it = 0; it < nsteps; ++it) {
    wait_async0();
    __syncthreads();
    int buf = it & 1;
    if (it + 1 < nsteps) {
      int ks = (it + 1) * 32;
      async_b128(&At[buf ^ 1][ldsOff], aBase + ks);
      async_b128(&At[buf ^ 1][ldsOff + 8], aBase + ks + 8);
      async_b128(&Bt[buf ^ 1][ldsOff], wBase + ks);
      async_b128(&Bt[buf ^ 1][ldsOff + 8], wBase + ks + 8);
      if (it + 2 < nsteps) {
        __builtin_prefetch(aBase + ks + 32, 0, 1);
        __builtin_prefetch(wBase + ks + 32, 0, 1);
      }
    }
    v16bf a0 = load_A_rm(&At[buf][(mb + 0) * 16 * 32], 32, lane);
    v16bf a1 = load_A_rm(&At[buf][(mb + 1) * 16 * 32], 32, lane);
#pragma unroll
    for (int j = 0; j < 4; ++j) {
      v16bf b = load_B_ct(&Bt[buf][(nb + j) * 16 * 32], 32, lane);
      acc[0][j] = wmma_bf(a0, b, acc[0][j]);
      acc[1][j] = wmma_bf(a1, b, acc[1][j]);
    }
  }
  // epilogue
#pragma unroll
  for (int m = 0; m < 2; ++m) {
    int r0 = bm * 128 + (mb + m) * 16 + (lane >> 4) * 8;
#pragma unroll
    for (int j = 0; j < 4; ++j) {
      int col = bn * 128 + (nb + j) * 16 + (lane & 15);
      float bv = bias ? bias[col] : 0.f;
#pragma unroll
      for (int r = 0; r < 8; ++r) {
        float v = acc[m][j][r] + bv;
        if (act) v = silu_f(v);
        size_t idx = (size_t)(r0 + r) * N + col;
        if (residual) v += residual[idx];
        if (outF) outF[idx] = v;
        if (outB) outB[idx] = f2bf(v);
      }
    }
  }
}

// ---------------- causal depthwise conv (K=4) + SiLU + gate multiply -----------------------
__global__ __launch_bounds__(256) void conv_gate_kernel(const __bf16* __restrict__ hup,
                                                        const float* __restrict__ cw,
                                                        const float* __restrict__ cb,
                                                        const __bf16* __restrict__ g,
                                                        __bf16* __restrict__ hg) {
  size_t i = (size_t)blockIdx.x * 256 + threadIdx.x;   // over B*L*DI
  int c = (int)(i % DDI);
  size_t bl = i / DDI;
  int l = (int)(bl % LL);
  float acc = cb[c];
#pragma unroll
  for (int j = 0; j < 4; ++j) {
    int ls = l - 3 + j;
    if (ls >= 0) acc += cw[c * 4 + j] * bf2f(hup[i + ((size_t)(ls - l)) * DDI]);
  }
  hg[i] = f2bf(silu_f(acc) * bf2f(g[i]));
}

// ---------------- chunked Hebbian scan ------------------------------------------------------
// grid = (B, D/64). Each workgroup owns M[:, e0:e0+64] resident in LDS (fp32, 256KB) for the
// whole sequence. Chunk C=64:
//   Y      = diag(d^{t+1}) * Q*M  +  (mask(t,s) d^{t+1-64} * (Q K^T)) * Vs
//   M_new  = d^64 * M + K^T * Vs            with  Vs[s] = d^{63-s} * v_s
// Q/K slices double-buffered via async global->LDS; kT slices async from pre-transposed k.
__global__ __launch_bounds__(256) void scan_kernel(const __bf16* __restrict__ q,
                                                   const __bf16* __restrict__ k,
                                                   const __bf16* __restrict__ kTr,
                                                   const __bf16* __restrict__ vTr,
                                                   __bf16* __restrict__ ys,
                                                   const float* __restrict__ decay_p) {
  extern __shared__ char smem[];
  float*  M_T = (float*)smem;                           // [64 n][1024 d]   256 KB
  float*  S   = (float*)(smem + 262144);                // [64 t][64 s]      16 KB
  __bf16* VsT = (__bf16*)(smem + 278528);               // [64 j][64 s]       8 KB
  __bf16* qt  = (__bf16*)(smem + 286720);               // 2 x [64 t][32 k]   8 KB
  __bf16* kt  = (__bf16*)(smem + 294912);               // 2 x [64 s][32 k]   8 KB
  __bf16* kT2 = (__bf16*)(smem + 303104);               // [64 d][64 s]       8 KB
  float*  powA = (float*)(smem + 311296);               // d^{63-s}
  float*  powY = (float*)(smem + 311552);               // d^{t+1}
  float*  powS = (float*)(smem + 311808);               // d^{t+1-64}

  int b = blockIdx.x, e0 = blockIdx.y * 64;
  int tid = threadIdx.x, lane = tid & 31, wv = tid >> 5;
  int mt = wv & 3, np = wv >> 2;
  float dec = decay_p[0];
  float dC = __powf(dec, 64.f);

  const __bf16* qB   = q   + (size_t)b * LL * DD;
  const __bf16* kB   = k   + (size_t)b * LL * DD;
  const __bf16* kTrB = kTr + (size_t)b * DD * LL;
  const __bf16* vTrB = vTr + (size_t)b * DD * LL;
  __bf16* yB = ys + (size_t)b * LL * DD;

  int srow = tid >> 2, sk0 = (tid & 3) * 8;   // staging coords for qt/kt (16B per thread)

  for (int i = tid; i < 64 * 1024; i += 256) M_T[i] = 0.f;
  if (tid < 64) {
    powA[tid] = __powf(dec, (float)(63 - tid));
    powY[tid] = __powf(dec, (float)(tid + 1));
    powS[tid] = __powf(dec, (float)(tid + 1 - 64));
  }
  __syncthreads();
  v8f zero = {};

  for (int c = 0; c < LL / 64; ++c) {
    int t0 = c * 64;
    {  // stage pre-scaled V^T: VsT[j][s] = d^{63-s} * vT[e0+j][t0+s] (contiguous both sides)
      int jj = tid >> 2, vs0 = (tid & 3) * 16;
      const __bf16* vp = vTrB + (size_t)(e0 + jj) * LL + t0 + vs0;
#pragma unroll
      for (int i = 0; i < 16; ++i)
        VsT[jj * 64 + vs0 + i] = f2bf(bf2f(vp[i]) * powA[vs0 + i]);
    }
    // issue first async Q/K slice (buffer 0)
    async_b128(&qt[srow * 32 + sk0], qB + (size_t)(t0 + srow) * DD + sk0);
    async_b128(&kt[srow * 32 + sk0], kB + (size_t)(t0 + srow) * DD + sk0);

    v8f accY[2] = {zero, zero};   // Y tile 64x64: wave -> (mt, cols np*32 + j*16)
    v8f accS[2] = {zero, zero};   // S tile 64x64: same mapping

    for (int it = 0; it < DD / 32; ++it) {  // contraction over D for Q*M and Q*K^T
      wait_async0();
      __syncthreads();
      int buf = it & 1;
      if (it + 1 < DD / 32) {
        int ks = (it + 1) * 32;
        async_b128(&qt[(buf ^ 1) * 2048 + srow * 32 + sk0],
                   qB + (size_t)(t0 + srow) * DD + ks + sk0);
        async_b128(&kt[(buf ^ 1) * 2048 + srow * 32 + sk0],
                   kB + (size_t)(t0 + srow) * DD + ks + sk0);
      }
      int ks = it * 32;
      v16bf a = load_A_rm(&qt[buf * 2048 + mt * 16 * 32], 32, lane);
#pragma unroll
      for (int j = 0; j < 2; ++j) {
        v16bf bm = load_B_ct_f32(&M_T[(np * 32 + j * 16) * 1024 + ks], 1024, lane);
        accY[j] = wmma_bf(a, bm, accY[j]);
        v16bf bk = load_B_ct(&kt[buf * 2048 + (np * 32 + j * 16) * 32], 32, lane);
        accS[j] = wmma_bf(a, bk, accS[j]);
      }
    }

    {  // apply causal decay mask to S, scale inter-chunk Y rows by d^{t+1}
      int row0 = mt * 16 + (lane >> 4) * 8;
#pragma unroll
      for (int r = 0; r < 8; ++r) {
        int t = row0 + r;
        float fy = powY[t];
        float fs = powS[t];
#pragma unroll
        for (int j = 0; j < 2; ++j) {
          int cs = np * 32 + j * 16 + (lane & 15);
          S[t * 64 + cs] = (cs <= t) ? accS[j][r] * fs : 0.f;
        }
        accY[0][r] *= fy;
        accY[1][r] *= fy;
      }
    }
    __syncthreads();

    // intra-chunk: Y += S * Vs   (contraction over s, 2 K-steps)
#pragma unroll
    for (int ks2 = 0; ks2 < 2; ++ks2) {
      v16bf a2 = load_A_rm_f32(&S[(mt * 16) * 64 + ks2 * 32], 64, lane);
#pragma unroll
      for (int j = 0; j < 2; ++j) {
        v16bf b2 = load_B_ct(&VsT[(np * 32 + j * 16) * 64 + ks2 * 32], 64, lane);
        accY[j] = wmma_bf(a2, b2, accY[j]);
      }
    }
    {  // store Y chunk
      int r0 = mt * 16 + (lane >> 4) * 8;
#pragma unroll
      for (int j = 0; j < 2; ++j) {
        int cj = e0 + np * 32 + j * 16 + (lane & 15);
#pragma unroll
        for (int r = 0; r < 8; ++r)
          yB[(size_t)(t0 + r0 + r) * DD + cj] = f2bf(accY[j][r]);
      }
    }

    // state update: M[:, n] = d^64 * M + K^T * Vs, 64 d-rows per step
    for (int ds = 0; ds < DD; ds += 64) {
      __syncthreads();
      {  // async stage K^T slice: kT2[dloc][s] = kT[ds+dloc][t0+s] (contiguous)
        int dl = tid >> 2, ss0 = (tid & 3) * 16;
        const __bf16* kp = kTrB + (size_t)(ds + dl) * LL + t0 + ss0;
        async_b128(&kT2[dl * 64 + ss0], kp);
        async_b128(&kT2[dl * 64 + ss0 + 8], kp + 8);
      }
      wait_async0();
      __syncthreads();
      v8f accM[2] = {zero, zero};
#pragma unroll
      for (int ks2 = 0; ks2 < 2; ++ks2) {
        v16bf a3 = load_A_rm(&kT2[(mt * 16) * 64 + ks2 * 32], 64, lane);
#pragma unroll
        for (int j = 0; j < 2; ++j) {
          v16bf b3 = load_B_ct(&VsT[(np * 32 + j * 16) * 64 + ks2 * 32], 64, lane);
          accM[j] = wmma_bf(a3, b3, accM[j]);
        }
      }
#pragma unroll
      for (int j = 0; j < 2; ++j) {
        int n = np * 32 + j * 16 + (lane & 15);
        int d0 = ds + mt * 16 + (lane >> 4) * 8;
        float* mp = &M_T[n * 1024 + d0];
#pragma unroll
        for (int r = 0; r < 8; ++r) mp[r] = dC * mp[r] + accM[j][r];
      }
    }
    __syncthreads();
  }
}

// ---------------- host-side orchestration ---------------------------------------------------
extern "C" void kernel_launch(void* const* d_in, const int* in_sizes, int n_in,
                              void* d_out, int out_size, void* d_ws, size_t ws_size,
                              hipStream_t stream) {
  (void)in_sizes; (void)n_in; (void)out_size; (void)ws_size;
  const float* x      = (const float*)d_in[0];
  const float* norm_w = (const float*)d_in[1];
  const float* up_w   = (const float*)d_in[2];
  const float* up_b   = (const float*)d_in[3];
  const float* gate_w = (const float*)d_in[4];
  const float* gate_b = (const float*)d_in[5];
  const float* down_w = (const float*)d_in[6];
  const float* down_b = (const float*)d_in[7];
  const float* conv_w = (const float*)d_in[8];
  const float* conv_b = (const float*)d_in[9];
  const float* wq     = (const float*)d_in[10];
  const float* wk     = (const float*)d_in[11];
  const float* wv     = (const float*)d_in[12];
  const float* wo     = (const float*)d_in[13];
  const float* decay  = (const float*)d_in[14];

  __bf16* ws = (__bf16*)d_ws;
  size_t off = 0;
  __bf16* up_w_t   = ws + off; off += (size_t)DD * DDI;   // [DI][D]
  __bf16* gate_w_t = ws + off; off += (size_t)DD * DDI;   // [DI][D]
  __bf16* down_w_t = ws + off; off += (size_t)DDI * DD;   // [D][DI]
  __bf16* wq_t     = ws + off; off += (size_t)DD * DD;
  __bf16* wk_t     = ws + off; off += (size_t)DD * DD;
  __bf16* wv_t     = ws + off; off += (size_t)DD * DD;
  __bf16* wo_t     = ws + off; off += (size_t)DD * DD;
  __bf16* bufA     = ws + off; off += (size_t)MM * DD;    // normed -> out_bf -> ys
  __bf16* bufB     = ws + off; off += (size_t)MM * DDI;   // hup  -> q | kTr
  __bf16* bufC     = ws + off; off += (size_t)MM * DDI;   // gate -> k | vTr
  __bf16* bufD     = ws + off; off += (size_t)MM * DDI;   // hg   -> v
  __bf16* kTrBuf   = bufB + (size_t)MM * DD;              // second half of bufB
  __bf16* vTrBuf   = bufC + (size_t)MM * DD;              // second half of bufC

  // transposed bf16 weights: Wt[N][K]
  cvt_t_kernel<<<dim3(DDI / 32, DD / 32), 256, 0, stream>>>(up_w, up_w_t, DD, DDI);
  cvt_t_kernel<<<dim3(DDI / 32, DD / 32), 256, 0, stream>>>(gate_w, gate_w_t, DD, DDI);
  cvt_t_kernel<<<dim3(DD / 32, DDI / 32), 256, 0, stream>>>(down_w, down_w_t, DDI, DD);
  cvt_t_kernel<<<dim3(DD / 32, DD / 32), 256, 0, stream>>>(wq, wq_t, DD, DD);
  cvt_t_kernel<<<dim3(DD / 32, DD / 32), 256, 0, stream>>>(wk, wk_t, DD, DD);
  cvt_t_kernel<<<dim3(DD / 32, DD / 32), 256, 0, stream>>>(wv, wv_t, DD, DD);
  cvt_t_kernel<<<dim3(DD / 32, DD / 32), 256, 0, stream>>>(wo, wo_t, DD, DD);

  rmsnorm_kernel<<<MM, 256, 0, stream>>>(x, norm_w, bufA);

  gemm_bf16_kernel<<<dim3(MM / 128, DDI / 128), 256, 0, stream>>>(
      bufA, up_w_t, up_b, nullptr, nullptr, bufB, MM, DD, DDI, 0);
  gemm_bf16_kernel<<<dim3(MM / 128, DDI / 128), 256, 0, stream>>>(
      bufA, gate_w_t, gate_b, nullptr, nullptr, bufC, MM, DD, DDI, 1);

  conv_gate_kernel<<<(MM * DDI) / 256, 256, 0, stream>>>(bufB, conv_w, conv_b, bufC, bufD);

  gemm_bf16_kernel<<<dim3(MM / 128, DD / 128), 256, 0, stream>>>(
      bufD, down_w_t, down_b, nullptr, nullptr, bufA, MM, DDI, DD, 0);

  gemm_bf16_kernel<<<dim3(MM / 128, DD / 128), 256, 0, stream>>>(
      bufA, wq_t, nullptr, nullptr, nullptr, bufB, MM, DD, DD, 0);
  gemm_bf16_kernel<<<dim3(MM / 128, DD / 128), 256, 0, stream>>>(
      bufA, wk_t, nullptr, nullptr, nullptr, bufC, MM, DD, DD, 0);
  gemm_bf16_kernel<<<dim3(MM / 128, DD / 128), 256, 0, stream>>>(
      bufA, wv_t, nullptr, nullptr, nullptr, bufD, MM, DD, DD, 0);

  // per-batch transposes for the scan's column-major consumers (into unused buffer halves)
  transpose_bf16_kernel<<<dim3(DD / 32, LL / 32, BB), 256, 0, stream>>>(bufC, kTrBuf);
  transpose_bf16_kernel<<<dim3(DD / 32, LL / 32, BB), 256, 0, stream>>>(bufD, vTrBuf);

  // 305 KB dynamic LDS: full M e-tile resident per WGP + double-buffered Q/K slices + tables
  scan_kernel<<<dim3(BB, DD / 64), 256, 312064, stream>>>(bufB, bufC, kTrBuf, vTrBuf,
                                                          bufA, decay);

  gemm_bf16_kernel<<<dim3(MM / 128, DD / 128), 256, 0, stream>>>(
      bufA, wo_t, nullptr, x, (float*)d_out, nullptr, MM, DD, DD, 0);
}
